// RAGVisionKnowledge_43868795961503
// MI455X (gfx1250) — matrix operand order repaired and checked by reading, not compiled, not run on hardware
//
#include <hip/hip_runtime.h>
#include <math.h>

// ---------------------------------------------------------------------------
// Problem constants (from reference)
// ---------------------------------------------------------------------------
#define BQ     256      // batch
#define DV     256
#define DKB    512
#define NKB    250000
#define KTOP   16
#define NHEAD  8
#define HDIM   32
#define TILE_N 1024
#define SUB_N  128
#define NUM_TILES ((NKB + TILE_N - 1) / TILE_N)   // 245

// ---------------------------------------------------------------------------
// WMMA types / helpers (gfx1250, wave32)
// ---------------------------------------------------------------------------
typedef __attribute__((ext_vector_type(16))) __bf16 v16bf;
typedef __attribute__((ext_vector_type(8)))  float  v8f;
typedef __attribute__((ext_vector_type(4)))  float  v4f;

union Frag16 {
    uint4  u4[2];
    v16bf  v;
    __bf16 h[16];
};

__device__ __forceinline__ v8f wmma_bf16(v16bf a, v16bf b, v8f c) {
    return __builtin_amdgcn_wmma_f32_16x16x32_bf16(
        /*neg_a=*/false, a, /*neg_b=*/false, b,
        /*c_mod=*/(short)0, c, /*reuse_a=*/false, /*reuse_b=*/false);
}

// ---------------------------------------------------------------------------
// fp32 -> bf16 2-D convert (weights, activations, strided copies)
// ---------------------------------------------------------------------------
__global__ __launch_bounds__(256) void cvt2d_bf16(
    const float* __restrict__ src, __bf16* __restrict__ dst,
    int rows, int cols, int sld, int dld)
{
    int i = blockIdx.x * 256 + threadIdx.x;
    if (i < rows * cols) {
        int r = i / cols, c = i - r * cols;
        dst[(size_t)r * dld + c] = (__bf16)src[(size_t)r * sld + c];
    }
}

// ---------------------------------------------------------------------------
// Generic GEMM:  C[M,N] = act( A[M,K](bf16) @ W[N,K](bf16)^T + bias )
// Block tile 128(M) x 64(N), 256 threads = 8 waves, wave w owns M-rows
// [w*16, w*16+16) and 4 N-tiles of 16. K stepped by 32 (one WMMA).
// ---------------------------------------------------------------------------
template<int ACT, typename OutT>
__global__ __launch_bounds__(256) void gemm_bf16(
    const __bf16* __restrict__ A, int lda,
    const __bf16* __restrict__ W, int ldw,
    const float*  __restrict__ bias,
    OutT* __restrict__ C, int ldc,
    int M, int N, int K)
{
    __shared__ __align__(16) __bf16 aLds[128 * 40];   // 32 bf16 + 8 pad per row

    const int tid  = threadIdx.x;
    const int wave = tid >> 5;
    const int lane = tid & 31;
    const int m0   = blockIdx.x * 128;
    const int n0   = blockIdx.y * 64;
    const int ah   = lane >> 4;        // K-half selector for A/B fragments
    const int lan  = lane & 15;

    v8f acc[4] = {};

    for (int kc = 0; kc < K; kc += 32) {
        // ---- stage A chunk: 128 rows x 32 bf16 (64B/row), 2 threads/row ----
        {
            int row  = tid >> 1;
            int half = tid & 1;
            const uint4* src = reinterpret_cast<const uint4*>(
                A + (size_t)(m0 + row) * lda + kc + half * 16);
            uint4 d0 = src[0];
            uint4 d1 = src[1];
            uint4* dst = reinterpret_cast<uint4*>(&aLds[row * 40 + half * 16]);
            dst[0] = d0; dst[1] = d1;
        }
        __syncthreads();

        // ---- A fragment for this wave's 16-row strip ----
        Frag16 af;
        int am = wave * 16 + lan;
        af.u4[0] = *reinterpret_cast<const uint4*>(&aLds[am * 40 + ah * 16]);
        af.u4[1] = *reinterpret_cast<const uint4*>(&aLds[am * 40 + 8 + ah * 16]);

        // ---- 4 B fragments straight from global (W is [N,K] row-major) ----
#pragma unroll
        for (int nt = 0; nt < 4; ++nt) {
            int n = n0 + nt * 16 + lan;
            const uint4* wp = reinterpret_cast<const uint4*>(
                W + (size_t)n * ldw + kc + ah * 16);
            Frag16 bf;
            bf.u4[0] = wp[0];
            bf.u4[1] = wp[1];
            acc[nt] = wmma_bf16(af.v, bf.v, acc[nt]);
        }
        __syncthreads();
    }

    // ---- epilogue: C layout (VGPR r -> M = r + 8*(lane>=16), N = lane&15) --
    int rowhi = ah * 8;
#pragma unroll
    for (int nt = 0; nt < 4; ++nt) {
        int n = n0 + nt * 16 + lan;
        float bv = bias ? bias[n] : 0.0f;
#pragma unroll
        for (int r = 0; r < 8; ++r) {
            int m = m0 + wave * 16 + rowhi + r;
            float v = acc[nt][r] + bv;
            if (ACT) v = fmaxf(v, 0.0f);
            C[(size_t)m * ldc + n] = (OutT)v;
        }
    }
}

// ---------------------------------------------------------------------------
// Fused cosine-similarity + per-tile top-16.
// Block: 256 threads (8 waves), one tile of TILE_N KB rows vs all 256 queries.
// KB streamed once from HBM with non-temporal hints (fp32 -> bf16 in
// registers); row norms fused into the same pass. A-fragment LDS loads are
// software-pipelined one WMMA ahead so the wave waits on dscnt<=2, not 0.
// ---------------------------------------------------------------------------
__global__ __launch_bounds__(256) void sims_topk(
    const float*  __restrict__ kb,     // [NKB, 512] fp32
    const __bf16* __restrict__ qbf,    // [256, 512] bf16 (unnormalized query)
    float* __restrict__ candS,         // [256, numTiles, 16]
    int*   __restrict__ candI)
{
    __shared__ __align__(16) __bf16 qLds[256 * 40];     // 20 KB
    __shared__ float sLds[256 * 129];                   // 132 KB (pad->no conflicts)

    const int tid   = threadIdx.x;
    const int wave  = tid >> 5;
    const int lane  = tid & 31;
    const int lan   = lane & 15;
    const int ah    = lane >> 4;
    const int tile0 = blockIdx.x * TILE_N;

    float topS[KTOP];
    int   topI[KTOP];
#pragma unroll
    for (int i = 0; i < KTOP; ++i) { topS[i] = -1e30f; topI[i] = 0; }
    float kmin = -1e30f;

    for (int sub = 0; sub < TILE_N / SUB_N; ++sub) {
        const int n0    = tile0 + sub * SUB_N + wave * 16;
        const int nmine = n0 + lan;
        const bool valid = (nmine < NKB);
        const float* krow = kb + (size_t)(valid ? nmine : 0) * DKB;

        v8f acc[16] = {};
        float sumsq = 0.0f;

        for (int kc = 0; kc < DKB; kc += 32) {
            // stage query chunk: 256 rows x 32 bf16, one thread per row
            {
                const uint4* src = reinterpret_cast<const uint4*>(
                    qbf + (size_t)tid * DKB + kc);
                uint4* dst = reinterpret_cast<uint4*>(&qLds[tid * 40]);
                dst[0] = src[0]; dst[1] = src[1]; dst[2] = src[2]; dst[3] = src[3];
            }
            __syncthreads();

            // B fragment from the KB stream (non-temporal fp32 -> bf16),
            // with fused sum of squares for the row norm.
            Frag16 bf;
            if (valid) {
                if (kc + 32 < DKB)
                    __builtin_prefetch(krow + kc + 32 + ah * 16, 0, 0);
                union { v4f f4[4]; float f[16]; } kd;
                const v4f* kp = reinterpret_cast<const v4f*>(krow + kc + ah * 16);
                kd.f4[0] = __builtin_nontemporal_load(kp + 0);
                kd.f4[1] = __builtin_nontemporal_load(kp + 1);
                kd.f4[2] = __builtin_nontemporal_load(kp + 2);
                kd.f4[3] = __builtin_nontemporal_load(kp + 3);
#pragma unroll
                for (int i = 0; i < 16; ++i) {
                    float x = kd.f[i];
                    sumsq += x * x;
                    bf.h[i] = (__bf16)x;
                }
            } else {
#pragma unroll
                for (int i = 0; i < 16; ++i) bf.h[i] = (__bf16)0.0f;
            }

            // 16 M-tiles of queries vs this wave's 16 KB rows,
            // A-fragment LDS loads pipelined one tile ahead.
            Frag16 afA, afB;
            {
                int am = lan;   // m = 0
                afA.u4[0] = *reinterpret_cast<const uint4*>(&qLds[am * 40 + ah * 16]);
                afA.u4[1] = *reinterpret_cast<const uint4*>(&qLds[am * 40 + 8 + ah * 16]);
            }
#pragma unroll
            for (int m = 0; m < 16; ++m) {
                Frag16& cur = (m & 1) ? afB : afA;
                Frag16& nxt = (m & 1) ? afA : afB;
                if (m < 15) {
                    int am = (m + 1) * 16 + lan;
                    nxt.u4[0] = *reinterpret_cast<const uint4*>(&qLds[am * 40 + ah * 16]);
                    nxt.u4[1] = *reinterpret_cast<const uint4*>(&qLds[am * 40 + 8 + ah * 16]);
                }
                acc[m] = wmma_bf16(cur.v, bf.v, acc[m]);
            }
            __syncthreads();
        }

        // combine K-halves of sumsq (lane l and l^16 hold same KB row)
        float scale;
        {
            float other = __shfl_xor(sumsq, 16, 32);
            scale = valid ? 1.0f / (sqrtf(sumsq + other) + 1e-12f) : 0.0f;
        }

        // stage scaled scores: sLds[query][col], col in [0,128)
        const int col   = wave * 16 + lan;
        const int rowhi = ah * 8;
#pragma unroll
        for (int m = 0; m < 16; ++m) {
#pragma unroll
            for (int r = 0; r < 8; ++r) {
                int q = m * 16 + rowhi + r;
                sLds[q * 129 + col] = valid ? acc[m][r] * scale : -1e30f;
            }
        }
        __syncthreads();

        // per-thread (= per-query) running top-16 over this sub-tile
        const int base = tile0 + sub * SUB_N;
        for (int c = 0; c < SUB_N; ++c) {
            float v = sLds[tid * 129 + c];
            if (v > kmin) {
                int gi = base + c;
                int p = KTOP - 1;
                while (p > 0 && topS[p - 1] < v) {
                    topS[p] = topS[p - 1]; topI[p] = topI[p - 1]; --p;
                }
                topS[p] = v; topI[p] = gi;
                kmin = topS[KTOP - 1];
            }
        }
        __syncthreads();
    }

    // candidates are written once, read once -> non-temporal stores
    size_t off = ((size_t)tid * gridDim.x + blockIdx.x) * KTOP;
#pragma unroll
    for (int i = 0; i < KTOP; ++i) {
        __builtin_nontemporal_store(topS[i], &candS[off + i]);
        __builtin_nontemporal_store(topI[i], &candI[off + i]);
    }
}

// ---------------------------------------------------------------------------
// Merge per-tile candidates -> final top-16 per query, then gather the
// selected KB rows converted to bf16 for the knowledge projector.
// One block per query.
// ---------------------------------------------------------------------------
__global__ __launch_bounds__(256) void topk_merge_gather(
    const float* __restrict__ candS, const int* __restrict__ candI,
    const float* __restrict__ kb,
    __bf16* __restrict__ retr_bf,      // [256*16, 512]
    int numTiles)
{
    __shared__ float s[NUM_TILES * KTOP];
    __shared__ int   si[NUM_TILES * KTOP];
    __shared__ float rmax[256];
    __shared__ int   rarg[256];
    __shared__ int   chosen[KTOP];

    const int q   = blockIdx.x;
    const int tid = threadIdx.x;
    const int nc  = numTiles * KTOP;

    for (int i = tid; i < nc; i += 256) {
        s[i]  = __builtin_nontemporal_load(&candS[(size_t)q * nc + i]);
        si[i] = __builtin_nontemporal_load(&candI[(size_t)q * nc + i]);
    }
    __syncthreads();

    for (int j = 0; j < KTOP; ++j) {
        float m = -2e30f; int a = 0;
        for (int i = tid; i < nc; i += 256)
            if (s[i] > m) { m = s[i]; a = i; }
        rmax[tid] = m; rarg[tid] = a;
        __syncthreads();
        for (int st = 128; st > 0; st >>= 1) {
            if (tid < st && rmax[tid + st] > rmax[tid]) {
                rmax[tid] = rmax[tid + st]; rarg[tid] = rarg[tid + st];
            }
            __syncthreads();
        }
        if (tid == 0) { int a0 = rarg[0]; chosen[j] = si[a0]; s[a0] = -3e30f; }
        __syncthreads();
    }

    // gather 16 rows x 512 fp32 -> bf16 (sparse one-shot rows: non-temporal)
    for (int e = tid; e < KTOP * DKB; e += 256) {
        int j = e >> 9, d = e & 511;
        float x = __builtin_nontemporal_load(&kb[(size_t)chosen[j] * DKB + d]);
        retr_bf[((size_t)q * KTOP + j) * DKB + d] = (__bf16)x;
    }
}

// ---------------------------------------------------------------------------
// Cross-attention: one block per batch element. q:[256], k/v:[16,256],
// 8 heads x 32 dims, softmax over the 16 retrieved entries.
// ---------------------------------------------------------------------------
__global__ __launch_bounds__(256) void attention(
    const float* __restrict__ qa,   // [256,256]
    const float* __restrict__ ka,   // [4096,256]
    const float* __restrict__ va,   // [4096,256]
    __bf16* __restrict__ ctx_bf)    // [256,256]
{
    __shared__ float qs[256];
    __shared__ float sc[NHEAD][KTOP];

    const int b = blockIdx.x, tid = threadIdx.x;
    qs[tid] = qa[(size_t)b * DV + tid];
    __syncthreads();

    if (tid < NHEAD * KTOP) {
        int h = tid >> 4, j = tid & 15;
        const float* kr = ka + ((size_t)b * KTOP + j) * DV + h * HDIM;
        float sv = 0.0f;
#pragma unroll
        for (int i = 0; i < HDIM; ++i) sv += qs[h * HDIM + i] * kr[i];
        sc[h][j] = sv * 0.17677669529663687f;   // 1/sqrt(32)
    }
    __syncthreads();

    if (tid < NHEAD) {
        int h = tid;
        float mx = -1e30f;
        for (int j = 0; j < KTOP; ++j) mx = fmaxf(mx, sc[h][j]);
        float sum = 0.0f;
        for (int j = 0; j < KTOP; ++j) { float e = expf(sc[h][j] - mx); sc[h][j] = e; sum += e; }
        float inv = 1.0f / sum;
        for (int j = 0; j < KTOP; ++j) sc[h][j] *= inv;
    }
    __syncthreads();

    const int d = tid, h = d >> 5;
    float c = 0.0f;
#pragma unroll
    for (int j = 0; j < KTOP; ++j)
        c += sc[h][j] * va[((size_t)b * KTOP + j) * DV + d];
    ctx_bf[(size_t)b * DV + d] = (__bf16)c;
}

// ---------------------------------------------------------------------------
// Residual + LayerNorm: one block per row of 256.
// ---------------------------------------------------------------------------
__global__ __launch_bounds__(256) void resid_ln(
    const float* __restrict__ vis, const float* __restrict__ op,
    const float* __restrict__ g, const float* __restrict__ be,
    float* __restrict__ out)
{
    __shared__ float red[256];
    const int b = blockIdx.x, t = threadIdx.x;
    float x = vis[(size_t)b * DV + t] + op[(size_t)b * DV + t];
    red[t] = x; __syncthreads();
    for (int st = 128; st > 0; st >>= 1) { if (t < st) red[t] += red[t + st]; __syncthreads(); }
    float mu = red[0] / 256.0f;
    __syncthreads();
    float d = x - mu;
    red[t] = d * d; __syncthreads();
    for (int st = 128; st > 0; st >>= 1) { if (t < st) red[t] += red[t + st]; __syncthreads(); }
    float var = red[0] / 256.0f;
    out[(size_t)b * DV + t] = d * rsqrtf(var + 1e-5f) * g[t] + be[t];
}

// ---------------------------------------------------------------------------
// Host orchestration
// ---------------------------------------------------------------------------
extern "C" void kernel_launch(void* const* d_in, const int* in_sizes, int n_in,
                              void* d_out, int out_size, void* d_ws, size_t ws_size,
                              hipStream_t stream) {
    const float* visual = (const float*)d_in[0];
    const float* kb     = (const float*)d_in[1];
    const float* qp_w1  = (const float*)d_in[2];  const float* qp_b1 = (const float*)d_in[3];
    const float* qp_w2  = (const float*)d_in[4];  const float* qp_b2 = (const float*)d_in[5];
    const float* kp_w1  = (const float*)d_in[6];  const float* kp_b1 = (const float*)d_in[7];
    const float* kp_w2  = (const float*)d_in[8];  const float* kp_b2 = (const float*)d_in[9];
    const float* wq     = (const float*)d_in[10]; const float* bq    = (const float*)d_in[11];
    const float* wk     = (const float*)d_in[12]; const float* bk    = (const float*)d_in[13];
    const float* wv     = (const float*)d_in[14]; const float* bv    = (const float*)d_in[15];
    const float* wo     = (const float*)d_in[16]; const float* bo    = (const float*)d_in[17];
    const float* out_w  = (const float*)d_in[18]; const float* out_b = (const float*)d_in[19];
    const float* ln_g   = (const float*)d_in[20]; const float* ln_b  = (const float*)d_in[21];
    float* out = (float*)d_out;

    char* ws = (char*)d_ws;
    size_t off = 0;
    auto alloc = [&](size_t bytes) -> char* {
        off = (off + 255) & ~(size_t)255;
        char* p = ws + off;
        off += bytes;
        return p;
    };

    // bf16 weight copies
    __bf16* w_qp1  = (__bf16*)alloc((size_t)512 * 256 * 2);
    __bf16* w_qp2  = (__bf16*)alloc((size_t)512 * 512 * 2);
    __bf16* w_kp1  = (__bf16*)alloc((size_t)512 * 512 * 2);
    __bf16* w_kp2  = (__bf16*)alloc((size_t)256 * 512 * 2);
    __bf16* w_q    = (__bf16*)alloc((size_t)256 * 256 * 2);
    __bf16* w_k    = (__bf16*)alloc((size_t)256 * 256 * 2);
    __bf16* w_v    = (__bf16*)alloc((size_t)256 * 256 * 2);
    __bf16* w_o    = (__bf16*)alloc((size_t)256 * 256 * 2);
    __bf16* w_out  = (__bf16*)alloc((size_t)256 * 512 * 2);
    // activations
    __bf16* vis_bf = (__bf16*)alloc((size_t)BQ * DV * 2);
    __bf16* h1     = (__bf16*)alloc((size_t)BQ * 512 * 2);
    __bf16* q_bf   = (__bf16*)alloc((size_t)BQ * DKB * 2);
    float*  candS  = (float*) alloc((size_t)BQ * NUM_TILES * KTOP * 4);
    int*    candI  = (int*)   alloc((size_t)BQ * NUM_TILES * KTOP * 4);
    __bf16* retr   = (__bf16*)alloc((size_t)BQ * KTOP * DKB * 2);
    __bf16* h2     = (__bf16*)alloc((size_t)BQ * KTOP * 512 * 2);
    __bf16* know   = (__bf16*)alloc((size_t)BQ * KTOP * DV * 2);
    float*  qa     = (float*) alloc((size_t)BQ * DV * 4);
    float*  ka     = (float*) alloc((size_t)BQ * KTOP * DV * 4);
    float*  va     = (float*) alloc((size_t)BQ * KTOP * DV * 4);
    __bf16* ctx_bf = (__bf16*)alloc((size_t)BQ * DV * 2);
    __bf16* comb   = (__bf16*)alloc((size_t)BQ * 2 * DV * 2);   // [256,512]
    float*  outp   = (float*) alloc((size_t)BQ * DV * 4);

    auto cvt = [&](const float* s, __bf16* d, int r, int c, int sld, int dld) {
        int n = r * c;
        cvt2d_bf16<<<(n + 255) / 256, 256, 0, stream>>>(s, d, r, c, sld, dld);
    };

    // weight + activation conversions
    cvt(qp_w1, w_qp1, 512, 256, 256, 256);
    cvt(qp_w2, w_qp2, 512, 512, 512, 512);
    cvt(kp_w1, w_kp1, 512, 512, 512, 512);
    cvt(kp_w2, w_kp2, 256, 512, 512, 512);
    cvt(wq,    w_q,   256, 256, 256, 256);
    cvt(wk,    w_k,   256, 256, 256, 256);
    cvt(wv,    w_v,   256, 256, 256, 256);
    cvt(wo,    w_o,   256, 256, 256, 256);
    cvt(out_w, w_out, 256, 512, 512, 512);
    cvt(visual, vis_bf, 256, 256, 256, 256);
    cvt(visual, comb,   256, 256, 256, 512);   // combined[:, 0:256] = visual

    // 1) query projector MLP
    gemm_bf16<1, __bf16><<<dim3(2, 8), 256, 0, stream>>>(
        vis_bf, 256, w_qp1, 256, qp_b1, h1, 512, 256, 512, 256);
    gemm_bf16<0, __bf16><<<dim3(2, 8), 256, 0, stream>>>(
        h1, 512, w_qp2, 512, qp_b2, q_bf, 512, 256, 512, 512);

    // 2) cosine-similarity retrieval (KB streamed once, NT) + top-k
    sims_topk<<<NUM_TILES, 256, 0, stream>>>(kb, q_bf, candS, candI);
    topk_merge_gather<<<BQ, 256, 0, stream>>>(candS, candI, kb, retr, NUM_TILES);

    // 3) knowledge projector MLP on [256*16, 512]
    gemm_bf16<1, __bf16><<<dim3(32, 8), 256, 0, stream>>>(
        retr, 512, w_kp1, 512, kp_b1, h2, 512, 4096, 512, 512);
    gemm_bf16<0, __bf16><<<dim3(32, 4), 256, 0, stream>>>(
        h2, 512, w_kp2, 512, kp_b2, know, 256, 4096, 256, 512);

    // 4) attention projections + attention
    gemm_bf16<0, float><<<dim3(2, 4), 256, 0, stream>>>(
        vis_bf, 256, w_q, 256, bq, qa, 256, 256, 256, 256);
    gemm_bf16<0, float><<<dim3(32, 4), 256, 0, stream>>>(
        know, 256, w_k, 256, bk, ka, 256, 4096, 256, 256);
    gemm_bf16<0, float><<<dim3(32, 4), 256, 0, stream>>>(
        know, 256, w_v, 256, bv, va, 256, 4096, 256, 256);
    attention<<<BQ, 256, 0, stream>>>(qa, ka, va, ctx_bf);
    gemm_bf16<0, __bf16><<<dim3(2, 4), 256, 0, stream>>>(
        ctx_bf, 256, w_o, 256, bo, comb + 256, 512, 256, 256, 256); // combined[:,256:512]

    // 5) output projection + residual LayerNorm
    gemm_bf16<0, float><<<dim3(2, 4), 256, 0, stream>>>(
        comb, 512, w_out, 512, out_b, outp, 256, 256, 256, 512);
    resid_ln<<<BQ, 256, 0, stream>>>(visual, outp, ln_g, ln_b, out);
}
